// Rwkv4NeoTimeMix_61581241090299
// MI455X (gfx1250) — matrix operand order repaired and checked by reading, not compiled, not run on hardware
//
#include <hip/hip_runtime.h>
#include <hip/hip_bf16.h>
#include <math.h>

// RWKV4 time-mix for MI455X (gfx1250, wave32, WMMA).
// B=4, T=2048, H=2048. GEMMs in bf16 via v_wmma_f32_16x16x32_bf16 (f32 accum).
// K-stage pipeline with GLOBAL_LOAD_ASYNC_TO_LDS_B128 (ASYNCcnt) feeding
// double-buffered LDS; one barrier per 64-wide K-stage; LDS rows padded to
// 72 bf16 so fragment ds_load_b128s are bank-conflict-free.
//
// Workspace: ~336 MB (bf16 weights + bf16 activations + f32 k/v).

#define Hdim 2048
#define Bdim 4
#define Tdim 2048
#define Mdim (Bdim * Tdim) /* 8192 */
#define LDST 72            /* padded LDS row stride in bf16 elements */

#if __has_builtin(__builtin_amdgcn_global_load_async_to_lds_b128)
#define USE_ASYNC_LDS 1
#else
#define USE_ASYNC_LDS 0
#endif

#define AS1 __attribute__((address_space(1)))
#define AS3 __attribute__((address_space(3)))

typedef __attribute__((ext_vector_type(16))) __bf16 bf16x16;
typedef __attribute__((ext_vector_type(8)))  __bf16 bf16x8;
typedef __attribute__((ext_vector_type(4)))  __bf16 bf16x4;
typedef __attribute__((ext_vector_type(8)))  float  f32x8;
typedef __attribute__((ext_vector_type(4)))  int    v4i;

static __device__ __forceinline__ __bf16 tobf(float f) { return (__bf16)f; }

#if USE_ASYNC_LDS
static __device__ __forceinline__ void async_b128(const __bf16* g, __bf16* l) {
  __builtin_amdgcn_global_load_async_to_lds_b128((AS1 v4i*)g, (AS3 v4i*)l, 0,
                                                 0);
}
static __device__ __forceinline__ void wait_async0() {
#if __has_builtin(__builtin_amdgcn_s_wait_asynccnt)
  __builtin_amdgcn_s_wait_asynccnt(0);
#else
  asm volatile("s_wait_asynccnt 0x0" ::: "memory");
#endif
}
#endif

// ---------------------------------------------------------------------------
// f32 -> bf16 conversion (weights)
// ---------------------------------------------------------------------------
__global__ __launch_bounds__(256) void cvt_bf16_kernel(
    const float* __restrict__ src, __bf16* __restrict__ dst) {
  const int i = (blockIdx.x * 256 + threadIdx.x) * 4;
  const float4 v = *(const float4*)&src[i];
  bf16x4 o;
  o.x = tobf(v.x); o.y = tobf(v.y); o.z = tobf(v.z); o.w = tobf(v.w);
  *(bf16x4*)&dst[i] = o;
}

// ---------------------------------------------------------------------------
// LayerNorm + token-shift mixing. One block per (b,t) row. Recomputes LN of
// the previous row locally (no inter-block dependency). Writes bf16 xk/xv/xr
// and x[:, -1, :] (f32) into d_out tail.
// ---------------------------------------------------------------------------
__global__ __launch_bounds__(256) void ln_mix_kernel(
    const float* __restrict__ hs, const float* __restrict__ lnw,
    const float* __restrict__ lnb, const float* __restrict__ mixk,
    const float* __restrict__ mixv, const float* __restrict__ mixr,
    __bf16* __restrict__ xk, __bf16* __restrict__ xv, __bf16* __restrict__ xr,
    float* __restrict__ xlast) {
  __shared__ float4 red[256];
  const int row = blockIdx.x;          // b*T + t
  const int t = row & (Tdim - 1);
  const int b = row >> 11;             // T = 2048
  const int tid = threadIdx.x;

  const float* xrow = hs + (size_t)row * Hdim;
  const float* prow = hs + (size_t)(row - 1) * Hdim;

  float xc[8], sc[8];
  float4 part = {0.f, 0.f, 0.f, 0.f};
#pragma unroll
  for (int i = 0; i < 8; ++i) {
    const int col = tid + i * 256;
    const float xv_ = xrow[col];
    const float sv_ = (t > 0) ? prow[col] : 0.f;
    xc[i] = xv_; sc[i] = sv_;
    part.x += xv_; part.y += xv_ * xv_;
    part.z += sv_; part.w += sv_ * sv_;
  }
  red[tid] = part;
  __syncthreads();
#pragma unroll
  for (int s = 128; s > 0; s >>= 1) {
    if (tid < s) {
      float4 o = red[tid + s];
      float4 m = red[tid];
      m.x += o.x; m.y += o.y; m.z += o.z; m.w += o.w;
      red[tid] = m;
    }
    __syncthreads();
  }
  const float4 tot = red[0];
  const float inv = 1.0f / (float)Hdim;
  const float mu  = tot.x * inv;
  const float var = tot.y * inv - mu * mu;
  const float rs  = rsqrtf(var + 1e-5f);
  const float mup  = tot.z * inv;
  const float varp = tot.w * inv - mup * mup;
  const float rsp  = rsqrtf(varp + 1e-5f);

#pragma unroll
  for (int i = 0; i < 8; ++i) {
    const int col = tid + i * 256;
    const float w = lnw[col], bb = lnb[col];
    const float xn = (xc[i] - mu) * rs * w + bb;
    const float sn = (t > 0) ? ((sc[i] - mup) * rsp * w + bb) : 0.f;
    const size_t off = (size_t)row * Hdim + col;
    xk[off] = tobf(sn + mixk[col] * (xn - sn));
    xv[off] = tobf(sn + mixv[col] * (xn - sn));
    xr[off] = tobf(sn + mixr[col] * (xn - sn));
    if (t == Tdim - 1) xlast[b * Hdim + col] = xn;
  }
}

// ---------------------------------------------------------------------------
// Tiled bf16 WMMA GEMM: D[m,n] = sum_k A[m,k] * W[n,k]  (both K-contiguous).
// Block tile 128x128, BK=64, 256 threads = 8 waves in a 4(M) x 2(N) grid,
// each wave owns 2x4 16x16 WMMA accumulators; 16 WMMAs per K-stage.
// mode 0: D -> f32 (k, v)
// mode 1: D -> sigmoid -> bf16 (r)
// mode 2: D -> resid + D -> f32 (final output)
// ---------------------------------------------------------------------------
__global__ __launch_bounds__(256, 1) void gemm_bf16_kernel(
    const __bf16* __restrict__ A, const __bf16* __restrict__ W,
    float* __restrict__ Df, __bf16* __restrict__ Db,
    const float* __restrict__ resid, int mode) {
  __shared__ __align__(16) __bf16 lds_a[2][128 * LDST];
  __shared__ __align__(16) __bf16 lds_b[2][128 * LDST];

  const int tid  = threadIdx.x;
  const int lane = tid & 31;
  const int wave = tid >> 5;
  const int wm = wave & 3;   // wave row (M)
  const int wn = wave >> 2;  // wave col (N)
  const int half = lane >> 4;
  const int l15  = lane & 15;
  const int blockN = blockIdx.x * 128;
  const int blockM = blockIdx.y * 128;

  // Staging chunk geometry: 128 rows x 64 k = 8192 bf16 per tile
  // = 1024 chunks of 8 bf16; 4 chunks per thread per tile (rows step 32).
  const int srow = tid >> 3;
  const int skp  = (tid & 7) << 3;

  // Uniform tile-base pointers (advance 64 elems per fetched stage) and
  // thread-invariant 32-bit offsets -> saddr+voffset global loads.
  const __bf16* aNext = A + (size_t)blockM * Hdim;
  const __bf16* wNext = W + (size_t)blockN * Hdim;
  int goff[4];
#pragma unroll
  for (int i = 0; i < 4; ++i) goff[i] = (srow + i * 32) * Hdim + skp;
  int loff[4];
#pragma unroll
  for (int i = 0; i < 4; ++i) loff[i] = (srow + i * 32) * LDST + skp;

  f32x8 acc[2][4];
#pragma unroll
  for (int mt = 0; mt < 2; ++mt)
#pragma unroll
    for (int nt = 0; nt < 4; ++nt) acc[mt][nt] = (f32x8)0.0f;

  const int nStage = Hdim / 64;  // 32

#if USE_ASYNC_LDS
  // Prologue: DMA tile 0 into buf0.
#pragma unroll
  for (int i = 0; i < 4; ++i) {
    async_b128(aNext + goff[i], &lds_a[0][loff[i]]);
    async_b128(wNext + goff[i], &lds_b[0][loff[i]]);
  }
  aNext += 64; wNext += 64;
  wait_async0();
  __syncthreads();
#else
  uint4 ar[4], br[4];
#pragma unroll
  for (int i = 0; i < 4; ++i) {
    ar[i] = *(const uint4*)(aNext + goff[i]);
    br[i] = *(const uint4*)(wNext + goff[i]);
  }
  aNext += 64; wNext += 64;
#pragma unroll
  for (int i = 0; i < 4; ++i) {
    *(uint4*)&lds_a[0][loff[i]] = ar[i];
    *(uint4*)&lds_b[0][loff[i]] = br[i];
  }
  __syncthreads();
#endif

  for (int it = 0; it < nStage; ++it) {
    const int cur = it & 1;
    const int nxt = 1 - cur;
    const bool more = (it + 1) < nStage;

#if USE_ASYNC_LDS
    // DMA tile it+1 into the other buffer; retires under this stage's WMMAs.
    // Safe: that buffer was last read before the barrier ending stage it-1.
    if (more) {
#pragma unroll
      for (int i = 0; i < 4; ++i) {
        async_b128(aNext + goff[i], &lds_a[nxt][loff[i]]);
        async_b128(wNext + goff[i], &lds_b[nxt][loff[i]]);
      }
      aNext += 64; wNext += 64;
    }
#else
    if (more) {
#pragma unroll
      for (int i = 0; i < 4; ++i) {
        ar[i] = *(const uint4*)(aNext + goff[i]);
        br[i] = *(const uint4*)(wNext + goff[i]);
      }
      aNext += 64; wNext += 64;
    }
#endif

    // Two 32-wide K-substeps from the current buffer.
#pragma unroll
    for (int ks = 0; ks < 2; ++ks) {
      // A fragment (16x32 bf16): lanes 0-15 hold K 0-7 & 16-23 of row M=lane,
      // lanes 16-31 hold K 8-15 & 24-31 (ISA 7.12.2 layout).
      bf16x16 afrag[2];
#pragma unroll
      for (int mt = 0; mt < 2; ++mt) {
        const int r = wm * 32 + mt * 16 + l15;
        const int base = r * LDST + ks * 32 + half * 8;
        const bf16x8 lo = *(const bf16x8*)&lds_a[cur][base];
        const bf16x8 hi = *(const bf16x8*)&lds_a[cur][base + 16];
        afrag[mt] = __builtin_shufflevector(lo, hi, 0, 1, 2, 3, 4, 5, 6, 7, 8,
                                            9, 10, 11, 12, 13, 14, 15);
      }
      // B fragment (32x16 bf16): lanes 0-15 hold K 0-7 & 16-23 of column N=lane,
      // lanes 16-31 hold K 8-15 & 24-31.
      bf16x16 bfrag[4];
#pragma unroll
      for (int nt = 0; nt < 4; ++nt) {
        const int n = wn * 64 + nt * 16 + l15;
        const int base = n * LDST + ks * 32 + half * 8;
        const bf16x8 lo = *(const bf16x8*)&lds_b[cur][base];
        const bf16x8 hi = *(const bf16x8*)&lds_b[cur][base + 16];
        bfrag[nt] = __builtin_shufflevector(lo, hi, 0, 1, 2, 3, 4, 5, 6, 7, 8,
                                            9, 10, 11, 12, 13, 14, 15);
      }
#pragma unroll
      for (int mt = 0; mt < 2; ++mt)
#pragma unroll
        for (int nt = 0; nt < 4; ++nt)
          acc[mt][nt] = __builtin_amdgcn_wmma_f32_16x16x32_bf16(
              false, afrag[mt], false, bfrag[nt], (short)0, acc[mt][nt], false,
              false);
    }

#if USE_ASYNC_LDS
    if (more) wait_async0();
#else
    if (more) {
#pragma unroll
      for (int i = 0; i < 4; ++i) {
        *(uint4*)&lds_a[nxt][loff[i]] = ar[i];
        *(uint4*)&lds_b[nxt][loff[i]] = br[i];
      }
    }
#endif
    __syncthreads();
  }

  // Epilogue. C/D layout: VGPR j = row j (lanes 0-15) / row 8+j (lanes 16-31).
#pragma unroll
  for (int mt = 0; mt < 2; ++mt)
#pragma unroll
    for (int nt = 0; nt < 4; ++nt) {
      const int rbase = blockM + wm * 32 + mt * 16 + half * 8;
      const int col   = blockN + wn * 64 + nt * 16 + l15;
#pragma unroll
      for (int j = 0; j < 8; ++j) {
        const size_t off = (size_t)(rbase + j) * Hdim + col;
        const float v = acc[mt][nt][j];
        if (mode == 0)
          Df[off] = v;
        else if (mode == 1)
          Db[off] = tobf(__builtin_amdgcn_rcpf(1.0f + __expf(-v)));
        else
          Df[off] = resid[off] + v;
      }
    }
}

// ---------------------------------------------------------------------------
// WKV max-stabilized recurrence: one thread per (b,h) channel, serial over T.
// Next-step k/v/r are prefetched before the transcendental chain so global
// latency overlaps compute. Division via v_rcp_f32. Fuses r * wkv and emits
// bf16 input for the output GEMM.
// ---------------------------------------------------------------------------
__global__ __launch_bounds__(256) void wkv_kernel(
    const float* __restrict__ kbuf, const float* __restrict__ vbuf,
    const __bf16* __restrict__ rbuf, __bf16* __restrict__ rw,
    const float* __restrict__ time_decay, const float* __restrict__ curr_w,
    float* __restrict__ numO, float* __restrict__ denO,
    float* __restrict__ normO) {
  const int idx = blockIdx.x * 256 + threadIdx.x;  // 0 .. B*H-1
  const int b = idx >> 11;                         // H = 2048
  const int h = idx & (Hdim - 1);

  const float decay = -__expf(time_decay[h]);
  const float cw = curr_w[h];

  float num = 0.f, den = 0.f, norm = -__builtin_inff();
  size_t off = (size_t)b * Tdim * Hdim + h;

  float kt = kbuf[off];
  float vt = vbuf[off];
  float rv = (float)rbuf[off];

  for (int t = 0; t < Tdim; ++t) {
    // Prefetch next step before the dependent exp chain.
    const size_t noff = off + Hdim;
    float kn = 0.f, vn = 0.f, rn = 0.f;
    if (t + 1 < Tdim) {
      kn = kbuf[noff];
      vn = vbuf[noff];
      rn = (float)rbuf[noff];
    }

    const float ct = cw + kt;
    const float cn = fmaxf(ct, norm);
    const float ep = __expf(norm - cn);
    const float ec = __expf(ct - cn);
    const float out =
        (ep * num + ec * vt) * __builtin_amdgcn_rcpf(ep * den + ec);
    rw[off] = tobf(rv * out);

    const float dpn = decay + norm;
    const float nn  = fmaxf(dpn, kt);
    const float ed  = __expf(dpn - nn);
    const float ek  = __expf(kt - nn);
    num  = ed * num + ek * vt;
    den  = ed * den + ek;
    norm = nn;

    kt = kn; vt = vn; rv = rn; off = noff;
  }
  numO[idx]  = num;
  denO[idx]  = den;
  normO[idx] = norm;
}

// ---------------------------------------------------------------------------
extern "C" void kernel_launch(void* const* d_in, const int* in_sizes, int n_in,
                              void* d_out, int out_size, void* d_ws,
                              size_t ws_size, hipStream_t stream) {
  const float* hs   = (const float*)d_in[0];
  const float* lnw  = (const float*)d_in[1];
  const float* lnb  = (const float*)d_in[2];
  const float* Wk   = (const float*)d_in[3];
  const float* Wv   = (const float*)d_in[4];
  const float* Wr   = (const float*)d_in[5];
  const float* Wo   = (const float*)d_in[6];
  const float* mixk = (const float*)d_in[7];
  const float* mixv = (const float*)d_in[8];
  const float* mixr = (const float*)d_in[9];
  const float* td   = (const float*)d_in[10];
  const float* cw   = (const float*)d_in[11];

  const size_t HH = (size_t)Hdim * Hdim;   // 4,194,304
  const size_t MH = (size_t)Mdim * Hdim;   // 16,777,216

  char* p = (char*)d_ws;
  __bf16* wkb = (__bf16*)p; p += HH * 2;
  __bf16* wvb = (__bf16*)p; p += HH * 2;
  __bf16* wrb = (__bf16*)p; p += HH * 2;
  __bf16* wob = (__bf16*)p; p += HH * 2;
  __bf16* xk  = (__bf16*)p; p += MH * 2;
  __bf16* xv  = (__bf16*)p; p += MH * 2;
  __bf16* xr  = (__bf16*)p; p += MH * 2;
  float*  kf  = (float*)p;  p += MH * 4;
  float*  vf  = (float*)p;  p += MH * 4;
  __bf16* rb  = (__bf16*)p; p += MH * 2;
  __bf16* rwb = (__bf16*)p; p += MH * 2;

  float* out   = (float*)d_out;
  float* xlast = out + MH;
  float* numO  = xlast + (size_t)Bdim * Hdim;
  float* denO  = numO + (size_t)Bdim * Hdim;
  float* normO = denO + (size_t)Bdim * Hdim;

  // 1. weights -> bf16
  const int cvtBlocks = (int)(HH / (256 * 4));  // 4096
  cvt_bf16_kernel<<<cvtBlocks, 256, 0, stream>>>(Wk, wkb);
  cvt_bf16_kernel<<<cvtBlocks, 256, 0, stream>>>(Wv, wvb);
  cvt_bf16_kernel<<<cvtBlocks, 256, 0, stream>>>(Wr, wrb);
  cvt_bf16_kernel<<<cvtBlocks, 256, 0, stream>>>(Wo, wob);

  // 2. LN + mixing
  ln_mix_kernel<<<Mdim, 256, 0, stream>>>(hs, lnw, lnb, mixk, mixv, mixr, xk,
                                          xv, xr, xlast);

  // 3. k, v, r GEMMs
  const dim3 ggrid(Hdim / 128, Mdim / 128);  // (16, 64)
  gemm_bf16_kernel<<<ggrid, 256, 0, stream>>>(xk, wkb, kf, nullptr, nullptr, 0);
  gemm_bf16_kernel<<<ggrid, 256, 0, stream>>>(xv, wvb, vf, nullptr, nullptr, 0);
  gemm_bf16_kernel<<<ggrid, 256, 0, stream>>>(xr, wrb, nullptr, rb, nullptr, 1);

  // 4. WKV scan (+ fuse r * wkv)
  wkv_kernel<<<(Bdim * Hdim) / 256, 256, 0, stream>>>(kf, vf, rb, rwb, td, cw,
                                                      numO, denO, normO);

  // 5. output GEMM + residual
  gemm_bf16_kernel<<<ggrid, 256, 0, stream>>>(rwb, wob, out, nullptr, hs, 2);
}
